// MomGatedSlotAttention_40407052321275
// MI455X (gfx1250) — matrix operand order repaired and checked by reading, not compiled, not run on hardware
//
#include <hip/hip_runtime.h>

// ---------------- problem constants (match reference) ----------------
constexpr int NB     = 2;      // batch
constexpr int NS     = 1024;   // seq
constexpr int NE     = 8;      // experts
constexpr int NTOPK  = 2;
constexpr int NHID   = 1024;
constexpr int NHEADS = 4;
constexpr int NSLOT  = 64;     // M slots
constexpr int NHK    = 256;    // head dim (k and v)
constexpr int NCAP   = 1024;   // capacity per (batch, expert)
constexpr int NCHUNK = 64;
constexpr int NTOK   = NB * NS;                 // 2048 tokens
constexpr int NROWS  = NE * NB * NCAP;          // 16384 expert-token rows
constexpr float EPSV      = 1e-5f;
constexpr float GATE_NORM = 8.0f;

typedef __attribute__((ext_vector_type(16))) __bf16          v16bf;
typedef __attribute__((ext_vector_type(8)))  float           v8f;
typedef __attribute__((ext_vector_type(2)))  float           v2f;
typedef __attribute__((ext_vector_type(16))) unsigned short  us16;
typedef __attribute__((ext_vector_type(8)))  unsigned short  us8;

static __device__ __forceinline__ unsigned short f2bf(float f) {
    unsigned u = __float_as_uint(f);
    unsigned r = u + 0x7FFFu + ((u >> 16) & 1u);   // round-to-nearest-even
    return (unsigned short)(r >> 16);
}

static __device__ __forceinline__ v8f wmma4(v8f acc, v2f a, v2f b) {
    return __builtin_amdgcn_wmma_f32_16x16x4_f32(false, a, false, b,
                                                 (short)0, acc, false, false);
}

static __device__ __forceinline__ v8f wmma32(v8f acc, v16bf a, v16bf b) {
    return __builtin_amdgcn_wmma_f32_16x16x32_bf16(false, a, false, b,
                                                   (short)0, acc, false, false);
}

// ---------------- f32 -> bf16 conversion (weights) ----------------
__global__ __launch_bounds__(256) void k_f2bf(const float* __restrict__ src,
                                              unsigned short* __restrict__ dst, long n) {
    long i = (long)blockIdx.x * blockDim.x + threadIdx.x;
    long stride = (long)gridDim.x * blockDim.x;
    for (; i < n; i += stride) dst[i] = f2bf(src[i]);
}

// ---------------- RMSNorm + router (softmax, top-2) ----------------
__global__ __launch_bounds__(256) void k_rmsnorm_gate(
        const float* __restrict__ x, const float* __restrict__ nw,
        const float* __restrict__ Wg,
        float* __restrict__ xn, unsigned short* __restrict__ xnb,
        int* __restrict__ sel, float* __restrict__ rw) {
    int t = blockIdx.x, tid = threadIdx.x;
    int wid = tid >> 5, lane = tid & 31;
    __shared__ float red[8];
    __shared__ float invn;
    __shared__ float scred[8][NE];
    __shared__ float score[NE];
    const float* xp = x + (size_t)t * NHID;
    float ss = 0.f;
    for (int i = tid; i < NHID; i += 256) { float v = xp[i]; ss += v * v; }
    for (int off = 16; off > 0; off >>= 1) ss += __shfl_down(ss, off, 32);
    if (lane == 0) red[wid] = ss;
    __syncthreads();
    if (tid == 0) {
        float s = 0.f;
        for (int w = 0; w < 8; ++w) s += red[w];
        invn = rsqrtf(s / (float)NHID + EPSV);
    }
    __syncthreads();
    float r = invn;
    float acc[NE];
#pragma unroll
    for (int e = 0; e < NE; ++e) acc[e] = 0.f;
    for (int i = tid; i < NHID; i += 256) {
        float y = xp[i] * r * nw[i];
        xn[(size_t)t * NHID + i] = y;
        xnb[(size_t)t * NHID + i] = f2bf(y);
#pragma unroll
        for (int e = 0; e < NE; ++e) acc[e] += y * Wg[e * NHID + i];
    }
#pragma unroll
    for (int e = 0; e < NE; ++e) {
        float a = acc[e];
        for (int off = 16; off > 0; off >>= 1) a += __shfl_down(a, off, 32);
        if (lane == 0) scred[wid][e] = a;
    }
    __syncthreads();
    if (tid < NE) {
        float s = 0.f;
        for (int w = 0; w < 8; ++w) s += scred[w][tid];
        score[tid] = s;
    }
    __syncthreads();
    if (tid == 0) {
        float mx = score[0];
        for (int e = 1; e < NE; ++e) mx = fmaxf(mx, score[e]);
        float ex[NE];
#pragma unroll
        for (int e = 0; e < NE; ++e) ex[e] = __expf(score[e] - mx);
        int i0 = 0; float v0 = ex[0];
        for (int e = 1; e < NE; ++e) if (ex[e] > v0) { v0 = ex[e]; i0 = e; }
        int i1 = -1; float v1 = -1.f;
        for (int e = 0; e < NE; ++e) if (e != i0 && ex[e] > v1) { v1 = ex[e]; i1 = e; }
        float denom = v0 + v1;
        sel[t * 2]     = i0; sel[t * 2 + 1] = i1;
        rw[t * 2]      = v0 / denom;
        rw[t * 2 + 1]  = v1 / denom;
    }
}

// ---------------- stable dispatch with capacity (last <=CAP kept) ----------------
__global__ void k_dispatch(const int* __restrict__ sel,
                           int* __restrict__ tok, int* __restrict__ inv) {
    int id = threadIdx.x;
    if (id >= NB * NE) return;
    int b = id / NE, e = id % NE;
    int cnt = 0;
    for (int s = 0; s < NS; ++s)
        for (int c = 0; c < NTOPK; ++c)
            if (sel[(b * NS + s) * NTOPK + c] == e) ++cnt;
    int idx = 0;
    for (int s = 0; s < NS; ++s)
        for (int c = 0; c < NTOPK; ++c) {
            int slot = (b * NS + s) * NTOPK + c;
            if (sel[slot] == e) {
                int p = idx + NCAP - cnt;          // left-pad; keep last <=CAP
                if (p >= 0 && p < NCAP) {
                    tok[(e * NB + b) * NCAP + p] = s;
                    inv[slot] = e * NCAP + p;
                } else {
                    inv[slot] = -1;
                }
                ++idx;
            }
        }
}

// ---------------- gather xe (bf16), zero padded rows ----------------
__global__ __launch_bounds__(256) void k_gather(const unsigned short* __restrict__ xnb,
                                                const int* __restrict__ tok,
                                                unsigned short* __restrict__ xe) {
    int row = blockIdx.x;                               // (e*NB+b)*NCAP + p
    int b = (row % (NB * NCAP)) / NCAP;
    int s = tok[row];
    const unsigned short* src = xnb + ((size_t)(b * NS + (s >= 0 ? s : 0))) * NHID;
    for (int i = threadIdx.x; i < NHID; i += 256)
        xe[(size_t)row * NHID + i] = (s >= 0) ? src[i] : (unsigned short)0;
}

// ---------------- bf16 WMMA GEMM: C[M,N] = A[M,K] * W[N,K]^T ----------------
// Register-tiled: each wave computes a 32x64 macro-tile (2 M-tiles x 4 N-tiles),
// 8 WMMAs per 12 b128 loads per K-step.
// mode 0: store f32; mode 1: swish; mode 2: gate (C=g, C2=slot)
__global__ __launch_bounds__(256) void k_gemm_bf16(
        const unsigned short* __restrict__ A,
        const unsigned short* __restrict__ W,
        float* __restrict__ C, float* __restrict__ C2,
        int M, int N, int K, int mode, long wstride) {
    int lane = threadIdx.x & 31;
    int wave = threadIdx.x >> 5;
    int m0 = blockIdx.x * 32;                  // two 16-row tiles
    int n0 = (blockIdx.y * 8 + wave) * 64;     // four 16-col tiles
    if (n0 >= N) return;
    const unsigned short* Wp = W;
    if (wstride) Wp += (size_t)(m0 / (NB * NCAP)) * wstride;   // per-expert weights

    int l16  = lane & 15;
    int asel = (lane >> 4) * 8;       // A half-lane K offset (0 or 8)
    int bsel = (lane >> 4) * 8;       // B half-lane K offset (0 or 8)
    const unsigned short* Ap0 = A  + (size_t)(m0 + l16) * K;
    const unsigned short* Ap1 = A  + (size_t)(m0 + 16 + l16) * K;
    const unsigned short* Bp  = Wp + (size_t)(n0 + l16) * K;

    v8f acc[2][4];
#pragma unroll
    for (int mi = 0; mi < 2; ++mi)
#pragma unroll
        for (int ni = 0; ni < 4; ++ni) acc[mi][ni] = (v8f){};

    for (int k0 = 0; k0 < K; k0 += 32) {
        __builtin_prefetch(Ap0 + k0 + 128, 0, 1);
        __builtin_prefetch(Ap1 + k0 + 128, 0, 1);
        us8 a0lo = *(const us8*)(Ap0 + k0 + asel);
        us8 a0hi = *(const us8*)(Ap0 + k0 + 16 + asel);
        us8 a1lo = *(const us8*)(Ap1 + k0 + asel);
        us8 a1hi = *(const us8*)(Ap1 + k0 + 16 + asel);
        v16bf a0 = __builtin_bit_cast(v16bf,
                __builtin_shufflevector(a0lo, a0hi, 0, 1, 2, 3, 4, 5, 6, 7,
                                                    8, 9, 10, 11, 12, 13, 14, 15));
        v16bf a1 = __builtin_bit_cast(v16bf,
                __builtin_shufflevector(a1lo, a1hi, 0, 1, 2, 3, 4, 5, 6, 7,
                                                    8, 9, 10, 11, 12, 13, 14, 15));
#pragma unroll
        for (int ni = 0; ni < 4; ++ni) {
            const unsigned short* bp = Bp + (size_t)ni * 16 * K + k0;
            __builtin_prefetch(bp + 128, 0, 1);
            us8 blo = *(const us8*)(bp + bsel);
            us8 bhi = *(const us8*)(bp + 16 + bsel);
            v16bf b = __builtin_bit_cast(v16bf,
                    __builtin_shufflevector(blo, bhi, 0, 1, 2, 3, 4, 5, 6, 7,
                                                      8, 9, 10, 11, 12, 13, 14, 15));
            acc[0][ni] = wmma32(acc[0][ni], a0, b);
            acc[1][ni] = wmma32(acc[1][ni], a1, b);
        }
    }

#pragma unroll
    for (int mi = 0; mi < 2; ++mi) {
#pragma unroll
        for (int ni = 0; ni < 4; ++ni) {
            int n = n0 + ni * 16 + l16;
            int mbase = m0 + mi * 16 + (lane >> 4) * 8;
#pragma unroll
            for (int r0 = 0; r0 < 8; ++r0) {
                float vv = acc[mi][ni][r0];
                int m = mbase + r0;
                if (mode == 1) {                                   // swish
                    vv = vv / (1.f + __expf(-vv));
                    C[(size_t)m * N + n] = vv;
                } else if (mode == 2) {                            // gate: g + slot
                    float g = (vv <= 0.f ? vv : 0.f) - log1pf(__expf(-fabsf(vv)));
                    g *= (1.0f / GATE_NORM);
                    C[(size_t)m * N + n]  = g;
                    C2[(size_t)m * N + n] = 1.f - __expf(g);
                } else {
                    C[(size_t)m * N + n] = vv;
                }
            }
        }
    }
}

// ---------------- GSA chunk scan: one block per (e,b,h), f32 WMMA ----------------
// LDS (320 KB): hk[256][64], hv[64][256], kt[64][256], qvt[64][256] (q then v),
//               Gt[64][64], si[64][64], Am[64][64] (A then qvg), qvT[64][64] (qv then Tm)
__global__ __launch_bounds__(256, 1) void k_gsa(
        const float* __restrict__ q, const float* __restrict__ k,
        const float* __restrict__ v, const float* __restrict__ sl,
        const float* __restrict__ gg, float* __restrict__ o) {
    int grp = blockIdx.x;                  // ((e*NB+b)*NHEADS + h)
    int h = grp % NHEADS, eb = grp / NHEADS;
    const size_t qoff = (size_t)eb * NCAP * NHID + (size_t)h * NHK;
    const size_t soff = (size_t)eb * NCAP * (NHEADS * NSLOT) + (size_t)h * NSLOT;
    const float* qg = q + qoff;
    const float* kg = k + qoff;
    const float* vg = v + qoff;
    const float* sg = sl + soff;
    const float* gG = gg + soff;
    float* og = o + qoff;

    extern __shared__ float sm[];
    float* hk  = sm;                  // [kd][m]
    float* hv  = hk  + 256 * 64;      // [m][dv]
    float* kt  = hv  + 64 * 256;      // [c][kd]
    float* qvt = kt  + 64 * 256;      // [c][d] : q chunk, later v chunk
    float* Gt  = qvt + 64 * 256;      // [c][m]
    float* si  = Gt  + 64 * 64;       // [c][m] s_in = slot*exp(-Gt)
    float* Am  = si  + 64 * 64;       // A then qvg
    float* qvT = Am  + 64 * 64;       // qv then Tm

    int tid = threadIdx.x;
    int lane = tid & 31, w = tid >> 5;
    int l16 = lane & 15;
    int kh  = (lane >> 4) * 2;        // K half offset for 16x16x4 fragments
    int rbb = (lane >> 4) * 8;        // C-frag row base

    for (int i = tid; i < 256 * 64; i += 256) hk[i] = 0.f;
    for (int i = tid; i < 64 * 256; i += 256) hv[i] = 0.f;
    __syncthreads();

    for (int ch = 0; ch < NCAP / NCHUNK; ++ch) {
        int t0 = ch * NCHUNK;
        // phase 1: gate cumsum + s_in; stage k and q chunks
        if (tid < NSLOT) {
            int m = tid; float run = 0.f;
            for (int c = 0; c < NCHUNK; ++c) {
                run += gG[(size_t)(t0 + c) * (NHEADS * NSLOT) + m];
                Gt[c * 64 + m] = run;
                si[c * 64 + m] = sg[(size_t)(t0 + c) * (NHEADS * NSLOT) + m] * __expf(-run);
            }
        }
        for (int i = tid; i < NCHUNK * NHK; i += 256) {
            int c = i >> 8, d = i & 255;
            kt[i]  = kg[(size_t)(t0 + c) * NHID + d];
            qvt[i] = qg[(size_t)(t0 + c) * NHID + d];
        }
        __syncthreads();

        // phase 2: A = tril(q k^T)  (16 tiles, K=256)
        for (int tt = w; tt < 16; tt += 8) {
            int it = tt >> 2, jt = tt & 3;
            v8f acc = {};
            const float* ar = qvt + (it * 16 + l16) * NHK;
            const float* br = kt  + (jt * 16 + l16) * NHK;
            for (int kk = 0; kk < NHK; kk += 4)
                acc = wmma4(acc, *(const v2f*)(ar + kk + kh), *(const v2f*)(br + kk + kh));
            int n = jt * 16 + l16;
#pragma unroll
            for (int r0 = 0; r0 < 8; ++r0) {
                int i = it * 16 + rbb + r0;
                Am[i * 64 + n] = (n <= i) ? acc[r0] : 0.f;
            }
        }
        __syncthreads();

        // phase 3: qv(pre) = exp(Gt) * (q@hk + A@s_in)  (16 tiles)
        for (int tt = w; tt < 16; tt += 8) {
            int it = tt >> 2, mt = tt & 3;
            int n = mt * 16 + l16;
            v8f acc = {};
            const float* ar = qvt + (it * 16 + l16) * NHK;
            for (int kk = 0; kk < NHK; kk += 4) {
                v2f bf; bf.x = hk[(kk + kh) * 64 + n]; bf.y = hk[(kk + kh + 1) * 64 + n];
                acc = wmma4(acc, *(const v2f*)(ar + kk + kh), bf);
            }
            const float* ar2 = Am + (it * 16 + l16) * 64;
            for (int kk = 0; kk < 64; kk += 4) {
                v2f bf; bf.x = si[(kk + kh) * 64 + n]; bf.y = si[(kk + kh + 1) * 64 + n];
                acc = wmma4(acc, *(const v2f*)(ar2 + kk + kh), bf);
            }
#pragma unroll
            for (int r0 = 0; r0 < 8; ++r0) {
                int i = it * 16 + rbb + r0;
                qvT[i * 64 + n] = __expf(Gt[i * 64 + n]) * acc[r0];
            }
        }
        __syncthreads();

        // phase 4: row softmax over slots
        if (tid < NCHUNK) {
            int i = tid;
            float mx = -3.4e38f;
            for (int m = 0; m < 64; ++m) mx = fmaxf(mx, qvT[i * 64 + m]);
            float s = 0.f;
            for (int m = 0; m < 64; ++m) {
                float e = __expf(qvT[i * 64 + m] - mx);
                qvT[i * 64 + m] = e; s += e;
            }
            float is = 1.f / s;
            for (int m = 0; m < 64; ++m) qvT[i * 64 + m] *= is;
        }
        __syncthreads();

        // phase 5: qvg -> Am
        for (int i = tid; i < NCHUNK * 64; i += 256)
            Am[i] = qvT[i] * __expf(Gt[i]);
        __syncthreads();

        // phase 6: stage v chunk (reuse q buffer) + Tm = tril(qvg @ s_in^T) (reuse qv buffer)
        for (int i = tid; i < NCHUNK * NHK; i += 256) {
            int c = i >> 8, d = i & 255;
            qvt[i] = vg[(size_t)(t0 + c) * NHID + d];
        }
        for (int tt = w; tt < 16; tt += 8) {
            int it = tt >> 2, jt = tt & 3;
            v8f acc = {};
            const float* ar = Am + (it * 16 + l16) * 64;
            const float* br = si + (jt * 16 + l16) * 64;
            for (int kk = 0; kk < 64; kk += 4)
                acc = wmma4(acc, *(const v2f*)(ar + kk + kh), *(const v2f*)(br + kk + kh));
            int n = jt * 16 + l16;
#pragma unroll
            for (int r0 = 0; r0 < 8; ++r0) {
                int i = it * 16 + rbb + r0;
                qvT[i * 64 + n] = (n <= i) ? acc[r0] : 0.f;
            }
        }
        __syncthreads();

        // phase 7: ov = Tm@v + qvg@hv -> global  (64 tiles)
        for (int tt = w; tt < 64; tt += 8) {
            int it = tt >> 4, nt = tt & 15;
            int n = nt * 16 + l16;
            v8f acc = {};
            const float* ar  = qvT + (it * 16 + l16) * 64;   // Tm
            const float* ar2 = Am  + (it * 16 + l16) * 64;   // qvg
            for (int kk = 0; kk < 64; kk += 4) {
                v2f bf; bf.x = qvt[(kk + kh) * 256 + n]; bf.y = qvt[(kk + kh + 1) * 256 + n];
                acc = wmma4(acc, *(const v2f*)(ar + kk + kh), bf);
                v2f bh; bh.x = hv[(kk + kh) * 256 + n]; bh.y = hv[(kk + kh + 1) * 256 + n];
                acc = wmma4(acc, *(const v2f*)(ar2 + kk + kh), bh);
            }
#pragma unroll
            for (int r0 = 0; r0 < 8; ++r0) {
                int i = it * 16 + rbb + r0;
                og[(size_t)(t0 + i) * NHID + n] = acc[r0];
            }
        }
        __syncthreads();

        // phase 8a: hk = gl * (hk + k^T @ s_in)   (64 tiles, disjoint ownership)
        for (int tt = w; tt < 64; tt += 8) {
            int it = tt >> 2, mt = tt & 3;
            int mrow = it * 16 + l16;          // kd
            int n = mt * 16 + l16;             // m
            v8f acc = {};
            for (int kk = 0; kk < 64; kk += 4) {
                v2f af; af.x = kt[(kk + kh) * 256 + mrow]; af.y = kt[(kk + kh + 1) * 256 + mrow];
                v2f bf; bf.x = si[(kk + kh) * 64 + n];     bf.y = si[(kk + kh + 1) * 64 + n];
                acc = wmma4(acc, af, bf);
            }
            float gl = __expf(Gt[63 * 64 + n]);
#pragma unroll
            for (int r0 = 0; r0 < 8; ++r0) {
                int kd = it * 16 + rbb + r0;
                hk[kd * 64 + n] = gl * (hk[kd * 64 + n] + acc[r0]);
            }
        }
        // phase 8b: hv = gl * (hv + s_in^T @ v)   (64 tiles, disjoint ownership)
        for (int tt = w; tt < 64; tt += 8) {
            int it = tt >> 4, nt = tt & 15;
            int mrow = it * 16 + l16;          // m
            int n = nt * 16 + l16;             // dv
            v8f acc = {};
            for (int kk = 0; kk < 64; kk += 4) {
                v2f af; af.x = si[(kk + kh) * 64 + mrow];  af.y = si[(kk + kh + 1) * 64 + mrow];
                v2f bf; bf.x = qvt[(kk + kh) * 256 + n];   bf.y = qvt[(kk + kh + 1) * 256 + n];
                acc = wmma4(acc, af, bf);
            }
#pragma unroll
            for (int r0 = 0; r0 < 8; ++r0) {
                int m = it * 16 + rbb + r0;
                float gl = __expf(Gt[63 * 64 + m]);
                hv[m * 256 + n] = gl * (hv[m * 256 + n] + acc[r0]);
            }
        }
        __syncthreads();
    }
}

// ---------------- combine: weighted gather via inverse map ----------------
__global__ __launch_bounds__(256) void k_combine(const float* __restrict__ o,
                                                 const int* __restrict__ inv,
                                                 const float* __restrict__ rw,
                                                 float* __restrict__ y) {
    int t = blockIdx.x;                       // b*NS + s
    int b = t / NS;
    int i0 = inv[t * 2], i1 = inv[t * 2 + 1];
    float w0 = rw[t * 2], w1 = rw[t * 2 + 1];
    size_t r0 = (i0 >= 0) ? ((size_t)(i0 / NCAP) * (NB * NCAP) + (size_t)b * NCAP + (i0 % NCAP)) * NHID : 0;
    size_t r1 = (i1 >= 0) ? ((size_t)(i1 / NCAP) * (NB * NCAP) + (size_t)b * NCAP + (i1 % NCAP)) * NHID : 0;
    for (int d = threadIdx.x; d < NHID; d += 256) {
        float acc = 0.f;
        if (i0 >= 0) acc += w0 * o[r0 + d];
        if (i1 >= 0) acc += w1 * o[r1 + d];
        y[(size_t)t * NHID + d] = acc;
    }
}

// ---------------- RMSNorm -> bf16 (pre output projection) ----------------
__global__ __launch_bounds__(256) void k_rmsnorm_bf(const float* __restrict__ x,
                                                    const float* __restrict__ nw,
                                                    unsigned short* __restrict__ yb) {
    int t = blockIdx.x, tid = threadIdx.x;
    int wid = tid >> 5, lane = tid & 31;
    __shared__ float red[8];
    __shared__ float invn;
    const float* xp = x + (size_t)t * NHID;
    float ss = 0.f;
    for (int i = tid; i < NHID; i += 256) { float v = xp[i]; ss += v * v; }
    for (int off = 16; off > 0; off >>= 1) ss += __shfl_down(ss, off, 32);
    if (lane == 0) red[wid] = ss;
    __syncthreads();
    if (tid == 0) {
        float s = 0.f;
        for (int w = 0; w < 8; ++w) s += red[w];
        invn = rsqrtf(s / (float)NHID + EPSV);
    }
    __syncthreads();
    float r = invn;
    for (int i = tid; i < NHID; i += 256)
        yb[(size_t)t * NHID + i] = f2bf(xp[i] * r * nw[i]);
}

// ---------------- host orchestration ----------------
extern "C" void kernel_launch(void* const* d_in, const int* in_sizes, int n_in,
                              void* d_out, int out_size, void* d_ws, size_t ws_size,
                              hipStream_t stream) {
    (void)in_sizes; (void)n_in; (void)out_size; (void)ws_size;
    const float* hs      = (const float*)d_in[0];
    const float* norm_w  = (const float*)d_in[1];
    const float* Wg      = (const float*)d_in[2];
    const float* Wq      = (const float*)d_in[3];
    const float* Wk      = (const float*)d_in[4];
    const float* Wv      = (const float*)d_in[5];
    const float* Wf      = (const float*)d_in[6];
    const float* gnorm_w = (const float*)d_in[7];
    const float* Wo      = (const float*)d_in[8];
    float* out = (float*)d_out;

    char* p = (char*)d_ws;
    auto alloc = [&](size_t bytes) -> void* {
        char* r = p;
        p += (bytes + 255) & ~(size_t)255;
        return (void*)r;
    };
    float*          xn  = (float*)alloc((size_t)NTOK * NHID * 4);
    unsigned short* xnb = (unsigned short*)alloc((size_t)NTOK * NHID * 2);
    int*            sel = (int*)alloc((size_t)NTOK * 2 * 4);
    float*          rw  = (float*)alloc((size_t)NTOK * 2 * 4);
    int*            inv = (int*)alloc((size_t)NTOK * 2 * 4);
    int*            tok = (int*)alloc((size_t)NROWS * 4);
    unsigned short* xe  = (unsigned short*)alloc((size_t)NROWS * NHID * 2);
    float*          qf  = (float*)alloc((size_t)NROWS * NHID * 4);
    float*          kf  = (float*)alloc((size_t)NROWS * NHID * 4);
    float*          vf  = (float*)alloc((size_t)NROWS * NHID * 4);
    float*          gf  = (float*)alloc((size_t)NROWS * 256 * 4);
    float*          sf  = (float*)alloc((size_t)NROWS * 256 * 4);
    float*          of  = (float*)alloc((size_t)NROWS * NHID * 4);
    float*          y   = (float*)alloc((size_t)NTOK * NHID * 4);
    unsigned short* yb  = (unsigned short*)alloc((size_t)NTOK * NHID * 2);
    unsigned short* Wqb = (unsigned short*)alloc((size_t)NHID * NHID * 2);
    unsigned short* Wkb = (unsigned short*)alloc((size_t)NE * NHID * NHID * 2);
    unsigned short* Wvb = (unsigned short*)alloc((size_t)NE * NHID * NHID * 2);
    unsigned short* Wfb = (unsigned short*)alloc((size_t)NE * 256 * NHID * 2);
    unsigned short* Wob = (unsigned short*)alloc((size_t)NHID * NHID * 2);

    // 1) weight conversions
    k_f2bf<<<2048, 256, 0, stream>>>(Wq, Wqb, (long)NHID * NHID);
    k_f2bf<<<2048, 256, 0, stream>>>(Wk, Wkb, (long)NE * NHID * NHID);
    k_f2bf<<<2048, 256, 0, stream>>>(Wv, Wvb, (long)NE * NHID * NHID);
    k_f2bf<<<2048, 256, 0, stream>>>(Wf, Wfb, (long)NE * 256 * NHID);
    k_f2bf<<<2048, 256, 0, stream>>>(Wo, Wob, (long)NHID * NHID);

    // 2) rmsnorm + router
    k_rmsnorm_gate<<<NTOK, 256, 0, stream>>>(hs, norm_w, Wg, xn, xnb, sel, rw);

    // 3) dispatch
    hipMemsetAsync(tok, 0xFF, (size_t)NROWS * 4, stream);
    k_dispatch<<<1, NB * NE, 0, stream>>>(sel, tok, inv);

    // 4) gather
    k_gather<<<NROWS, 256, 0, stream>>>(xnb, tok, xe);

    // 5) projections (WMMA bf16, 32x64 macro-tiles per wave)
    dim3 gq(NROWS / 32, NHID / 512);
    k_gemm_bf16<<<gq, 256, 0, stream>>>(xe, Wqb, qf, nullptr, NROWS, NHID, NHID, 1, 0);
    k_gemm_bf16<<<gq, 256, 0, stream>>>(xe, Wkb, kf, nullptr, NROWS, NHID, NHID, 1,
                                        (long)NHID * NHID);
    k_gemm_bf16<<<gq, 256, 0, stream>>>(xe, Wvb, vf, nullptr, NROWS, NHID, NHID, 0,
                                        (long)NHID * NHID);
    dim3 gfgrid(NROWS / 32, 1);
    k_gemm_bf16<<<gfgrid, 256, 0, stream>>>(xe, Wfb, gf, sf, NROWS, 256, NHID, 2,
                                            (long)256 * NHID);

    // 6) GSA chunk scan (320 KB LDS per workgroup, f32 WMMA)
    size_t smem = (size_t)(4 * 64 * 256 + 4 * 64 * 64) * sizeof(float);   // 327680 B
    k_gsa<<<NE * NB * NHEADS, 256, smem, stream>>>(qf, kf, vf, sf, gf, of);

    // 7) combine + final norm + output projection
    k_combine<<<NTOK, 256, 0, stream>>>(of, inv, rw, y);
    k_rmsnorm_bf<<<NTOK, 256, 0, stream>>>(y, gnorm_w, yb);
    dim3 go(NTOK / 32, NHID / 512);
    k_gemm_bf16<<<go, 256, 0, stream>>>(yb, Wob, out, nullptr, NTOK, NHID, NHID, 0, 0);
}
